// TransformerBlock_17059610100280
// MI455X (gfx1250) — compile-verified, matrix operand order checked by reading, not run on hardware
//
#include <hip/hip_runtime.h>
#include <hip/hip_bf16.h>
#include <math.h>
#include <stdint.h>

typedef unsigned short u16;
typedef unsigned int   u32;
typedef __attribute__((ext_vector_type(16))) __bf16 bf16x16;
typedef __attribute__((ext_vector_type(8)))  float  f32x8;
typedef __attribute__((ext_vector_type(4)))  u32    u32x4;

union Frag { bf16x16 v; u32x4 q[2]; };

static __device__ __forceinline__ u16 f2bf(float f) {
  u32 u = __float_as_uint(f);
  u = (u + 0x7fffu + ((u >> 16) & 1u)) >> 16;   // round-to-nearest-even
  return (u16)u;
}
static __device__ __forceinline__ float bf2f(u16 h) {
  return __uint_as_float(((u32)h) << 16);
}

// async copy of 32 contiguous bytes per lane: global -> LDS (CDNA5 ASYNCcnt path)
static __device__ __forceinline__ void async_cp32(u32 lds_off, const u16* gptr) {
  unsigned long long ga = (unsigned long long)(uintptr_t)gptr;
  asm volatile("global_load_async_to_lds_b128 %0, %1, off"
               :: "v"(lds_off), "v"(ga) : "memory");
  asm volatile("global_load_async_to_lds_b128 %0, %1, off offset:16"
               :: "v"(lds_off), "v"(ga) : "memory");
}
static __device__ __forceinline__ void wait_async0() {
  asm volatile("s_wait_asynccnt 0" ::: "memory");
}

// ---------------------------------------------------------------- convert
__global__ void cvt_f32_bf16(const float* __restrict__ in, u16* __restrict__ out, size_t n) {
  size_t i = (size_t)blockIdx.x * blockDim.x + threadIdx.x;
  size_t stride = (size_t)gridDim.x * blockDim.x;
  for (; i < n; i += stride) out[i] = f2bf(in[i]);
}

// ---------------------------------------------------------------- rmsnorm -> bf16
__global__ __launch_bounds__(256)
void rmsnorm_to_bf16(const float* __restrict__ x, const float* __restrict__ w,
                     u16* __restrict__ out, int D) {
  __shared__ float red[8];
  const int row = blockIdx.x;
  const float* xr = x + (size_t)row * D;
  float ss = 0.f;
  for (int c = threadIdx.x; c < D; c += 256) { float v = xr[c]; ss += v * v; }
  for (int off = 16; off > 0; off >>= 1) ss += __shfl_xor(ss, off, 32);
  if ((threadIdx.x & 31) == 0) red[threadIdx.x >> 5] = ss;
  __syncthreads();
  float tot = red[0]+red[1]+red[2]+red[3]+red[4]+red[5]+red[6]+red[7];
  float rinv = rsqrtf(tot / (float)D + 1e-6f);
  u16* orow = out + (size_t)row * D;
  for (int c = threadIdx.x; c < D; c += 256) orow[c] = f2bf(xr[c] * rinv * w[c]);
}

// ---------------------------------------------------------------- WMMA GEMM
// C[M,N] = A[M,K] * W[N,K]^T + bias, EPI: 0=bf16 out, 1=bf16 gelu, 2=f32 out + residual
// Double-buffered LDS filled by global_load_async_to_lds_b128 (ASYNCcnt),
// one barrier + one s_wait_asynccnt per 32-wide K-step.
template <int EPI>
__global__ __launch_bounds__(256, 2)
void gemm_bf16(const u16* __restrict__ A, const u16* __restrict__ Bw,
               const float* __restrict__ bias, const float* __restrict__ res,
               void* __restrict__ outp, int M, int N, int K) {
  __shared__ u16 As[2][128 * 40];
  __shared__ u16 Bs[2][128 * 40];
  const int tid  = threadIdx.x;
  const int wave = tid >> 5;
  const int lane = tid & 31;
  const int l16  = lane & 15;
  const int half = lane >> 4;
  const int bm0  = blockIdx.y * 128;
  const int bn0  = blockIdx.x * 128;
  const int wm0  = (wave & 3) * 32;
  const int wn0  = (wave >> 2) * 64;
  const int lrow = tid >> 1;
  const int lcol = (tid & 1) * 16;

  const u16* gA = A  + (size_t)(bm0 + lrow) * K + lcol;
  const u16* gB = Bw + (size_t)(bn0 + lrow) * K + lcol;
  const u32 lA[2] = { (u32)(uintptr_t)&As[0][lrow * 40 + lcol],
                      (u32)(uintptr_t)&As[1][lrow * 40 + lcol] };
  const u32 lB[2] = { (u32)(uintptr_t)&Bs[0][lrow * 40 + lcol],
                      (u32)(uintptr_t)&Bs[1][lrow * 40 + lcol] };

  const f32x8 zero = {0.f,0.f,0.f,0.f,0.f,0.f,0.f,0.f};
  f32x8 acc[2][4];
  #pragma unroll
  for (int i = 0; i < 2; ++i)
    #pragma unroll
    for (int j = 0; j < 4; ++j) acc[i][j] = zero;

  // prologue: async-fill buffer 0 with K-step 0
  async_cp32(lA[0], gA);
  async_cp32(lB[0], gB);

  const int nk = K >> 5;
  for (int t = 0; t < nk; ++t) {
    wait_async0();        // our buf[t&1] loads are the only outstanding ones
    __syncthreads();      // all waves' tiles visible; all waves done reading buf[t^1]

    if (t + 1 < nk) {     // prefetch next K-step into the other buffer
      const int nb = (t + 1) & 1;
      const u16* pa = gA + (size_t)(t + 1) * 32;
      const u16* pb = gB + (size_t)(t + 1) * 32;
      async_cp32(lA[nb], pa);
      async_cp32(lB[nb], pb);
    }

    const u16* as = &As[t & 1][0];
    const u16* bs = &Bs[t & 1][0];
    Frag af[2], bf[4];
    #pragma unroll
    for (int i = 0; i < 2; ++i) {
      const int r = (wm0 + 16 * i + l16) * 40;
      af[i].q[0] = *(const u32x4*)&as[r + half * 8];
      af[i].q[1] = *(const u32x4*)&as[r + 16 + half * 8];
    }
    #pragma unroll
    for (int j = 0; j < 4; ++j) {
      const int r = (wn0 + 16 * j + l16) * 40;
      bf[j].q[0] = *(const u32x4*)&bs[r + half * 8];
      bf[j].q[1] = *(const u32x4*)&bs[r + 16 + half * 8];
    }
    #pragma unroll
    for (int i = 0; i < 2; ++i)
      #pragma unroll
      for (int j = 0; j < 4; ++j)
        acc[i][j] = __builtin_amdgcn_wmma_f32_16x16x32_bf16(
            false, af[i].v, false, bf[j].v, (short)0, acc[i][j], false, false);
  }

  #pragma unroll
  for (int i = 0; i < 2; ++i) {
    #pragma unroll
    for (int j = 0; j < 4; ++j) {
      const int col = bn0 + wn0 + 16 * j + l16;
      const float bv = bias[col];
      #pragma unroll
      for (int r = 0; r < 8; ++r) {
        const int row = bm0 + wm0 + 16 * i + r + 8 * half;
        float v = acc[i][j][r] + bv;
        const size_t idx = (size_t)row * N + col;
        if (EPI == 0) {
          ((u16*)outp)[idx] = f2bf(v);
        } else if (EPI == 1) {
          float g = 0.5f * v * (1.0f + erff(v * 0.70710678118654752f));
          ((u16*)outp)[idx] = f2bf(g);
        } else {
          ((float*)outp)[idx] = res[idx] + v;
        }
      }
    }
  }
}

// ---------------------------------------------------------------- RoPE + split + V transpose
// qkv: [T, 6144] bf16 -> Q,K: [B,H,S,128] bf16 (rotated), Vt: [B,H,128,S] bf16
__global__ __launch_bounds__(256)
void rope_split(const u16* __restrict__ qkv, u16* __restrict__ Qo,
                u16* __restrict__ Ko, u16* __restrict__ Vt) {
  const int S = 2048;
  const int idx = blockIdx.x * 256 + threadIdx.x;   // t*2048 + h*128 + d
  const int t  = idx >> 11;
  const int hd = idx & 2047;
  const int h  = hd >> 7;
  const int d  = hd & 127;
  const int b  = t >> 11;       // t / S
  const int s  = t & 2047;      // t % S
  const size_t base = (size_t)t * 6144;

  const int i = d & 63;
  // inv_freq = 10000^(-2i/128) = exp2(-(log2(10000)/64) * i): single TRANS op
  const float inv = exp2f(-0.2076205059304601f * (float)i);
  const float ang = (float)s * inv;
  const float c = cosf(ang), sn = sinf(ang);

  const float q1 = bf2f(qkv[base + h * 128 + i]);
  const float q2 = bf2f(qkv[base + h * 128 + 64 + i]);
  const float k1 = bf2f(qkv[base + 2048 + h * 128 + i]);
  const float k2 = bf2f(qkv[base + 2048 + h * 128 + 64 + i]);
  const float qv = (d < 64) ? (q1 * c - q2 * sn) : (q1 * sn + q2 * c);
  const float kv = (d < 64) ? (k1 * c - k2 * sn) : (k1 * sn + k2 * c);
  const size_t od = ((size_t)((b * 16 + h) * S + s)) * 128 + d;
  Qo[od] = f2bf(qv);
  Ko[od] = f2bf(kv);
  Vt[((size_t)(b * 16 + h) * 128 + d) * S + s] = qkv[base + 4096 + h * 128 + d];
}

// ---------------------------------------------------------------- flash attention (1 wave per 16-row Q tile)
__global__ __launch_bounds__(256, 2)
void attn_flash(const u16* __restrict__ Q, const u16* __restrict__ Kc,
                const u16* __restrict__ Vt, u16* __restrict__ O) {
  __shared__ u16 Psh[8][16 * 32];
  const int S = 2048;
  const int tid = threadIdx.x;
  const int wave = tid >> 5, lane = tid & 31, l16 = lane & 15, half = lane >> 4;
  const int gw = blockIdx.x * 8 + wave;
  const int bh = gw >> 7;     // / (S/16)
  const int qt = gw & 127;
  const int q0 = qt * 16;

  const u16* Qb = Q  + (size_t)bh * S * 128;
  const u16* Kb = Kc + (size_t)bh * S * 128;
  const u16* Vb = Vt + (size_t)bh * 128 * S;
  u16* psh = &Psh[wave][0];

  Frag qa[4];
  #pragma unroll
  for (int c = 0; c < 4; ++c) {
    const u16* gp = Qb + (size_t)(q0 + l16) * 128 + c * 32;
    qa[c].q[0] = *(const u32x4*)(gp + half * 8);
    qa[c].q[1] = *(const u32x4*)(gp + 16 + half * 8);
  }

  const f32x8 zero = {0.f,0.f,0.f,0.f,0.f,0.f,0.f,0.f};
  f32x8 o[8];
  #pragma unroll
  for (int d = 0; d < 8; ++d) o[d] = zero;
  float m8[8], l8[8];
  #pragma unroll
  for (int r = 0; r < 8; ++r) { m8[r] = -1e30f; l8[r] = 0.f; }
  const float scale = 0.08838834764831845f;  // 1/sqrt(128)
  const int kbmax = (q0 + 15) >> 5;

  for (int kb = 0; kb <= kbmax; ++kb) {
    const int k0 = kb * 32;
    f32x8 s0 = zero, s1 = zero;
    #pragma unroll
    for (int c = 0; c < 4; ++c) {
      Frag kf0, kf1;
      const u16* g0 = Kb + (size_t)(k0 + l16) * 128 + c * 32;
      const u16* g1 = Kb + (size_t)(k0 + 16 + l16) * 128 + c * 32;
      kf0.q[0] = *(const u32x4*)(g0 + half * 8);
      kf0.q[1] = *(const u32x4*)(g0 + 16 + half * 8);
      kf1.q[0] = *(const u32x4*)(g1 + half * 8);
      kf1.q[1] = *(const u32x4*)(g1 + 16 + half * 8);
      s0 = __builtin_amdgcn_wmma_f32_16x16x32_bf16(false, qa[c].v, false, kf0.v, (short)0, s0, false, false);
      s1 = __builtin_amdgcn_wmma_f32_16x16x32_bf16(false, qa[c].v, false, kf1.v, (short)0, s1, false, false);
    }
    const int c0 = k0 + l16, c1 = c0 + 16;
    #pragma unroll
    for (int r = 0; r < 8; ++r) {
      const int row = q0 + r + 8 * half;
      float v0 = (c0 <= row) ? s0[r] * scale : -1e30f;
      float v1 = (c1 <= row) ? s1[r] * scale : -1e30f;
      float mx = fmaxf(v0, v1);
      mx = fmaxf(mx, __shfl_xor(mx, 1, 32));
      mx = fmaxf(mx, __shfl_xor(mx, 2, 32));
      mx = fmaxf(mx, __shfl_xor(mx, 4, 32));
      mx = fmaxf(mx, __shfl_xor(mx, 8, 32));
      const float mnew = fmaxf(m8[r], mx);
      const float al = __expf(m8[r] - mnew);
      m8[r] = mnew;
      const float e0 = __expf(v0 - mnew);
      const float e1 = __expf(v1 - mnew);
      float rs = e0 + e1;
      rs += __shfl_xor(rs, 1, 32);
      rs += __shfl_xor(rs, 2, 32);
      rs += __shfl_xor(rs, 4, 32);
      rs += __shfl_xor(rs, 8, 32);
      l8[r] = l8[r] * al + rs;
      const int prow = r + 8 * half;
      psh[prow * 32 + l16]      = f2bf(e0);
      psh[prow * 32 + 16 + l16] = f2bf(e1);
      #pragma unroll
      for (int d = 0; d < 8; ++d) o[d][r] *= al;
    }
    // C-layout -> A-layout transpose through per-wave LDS (in-order within wave)
    Frag pa;
    pa.q[0] = *(const u32x4*)&psh[l16 * 32 + half * 8];
    pa.q[1] = *(const u32x4*)&psh[l16 * 32 + 16 + half * 8];
    #pragma unroll
    for (int d = 0; d < 8; ++d) {
      Frag vf;
      const u16* gv = Vb + (size_t)(d * 16 + l16) * S + k0;
      vf.q[0] = *(const u32x4*)(gv + half * 8);
      vf.q[1] = *(const u32x4*)(gv + 16 + half * 8);
      o[d] = __builtin_amdgcn_wmma_f32_16x16x32_bf16(false, pa.v, false, vf.v, (short)0, o[d], false, false);
    }
  }

  const int b = bh >> 4, h = bh & 15;
  #pragma unroll
  for (int d = 0; d < 8; ++d) {
    const int dh = d * 16 + l16;
    #pragma unroll
    for (int r = 0; r < 8; ++r) {
      const int srow = q0 + r + 8 * half;
      O[(size_t)(b * S + srow) * 2048 + h * 128 + dh] = f2bf(o[d][r] / l8[r]);
    }
  }
}

// ---------------------------------------------------------------- host
extern "C" void kernel_launch(void* const* d_in, const int* in_sizes, int n_in,
                              void* d_out, int out_size, void* d_ws, size_t ws_size,
                              hipStream_t stream) {
  const float* x     = (const float*)d_in[0];
  const float* ln1_w = (const float*)d_in[1];
  const float* ln2_w = (const float*)d_in[2];
  const float* qkv_w = (const float*)d_in[3];
  const float* qkv_b = (const float*)d_in[4];
  const float* out_w = (const float*)d_in[5];
  const float* out_b = (const float*)d_in[6];
  const float* fc1_w = (const float*)d_in[7];
  const float* fc1_b = (const float*)d_in[8];
  const float* fc2_w = (const float*)d_in[9];
  const float* fc2_b = (const float*)d_in[10];
  (void)in_sizes; (void)n_in; (void)out_size; (void)ws_size;

  const int B = 2, S = 2048, D = 2048, DFF = 8192, T = B * S;

  char* p = (char*)d_ws;
  auto alloc = [&](size_t bytes) -> char* {
    char* r = p; p += (bytes + 255) & ~(size_t)255; return r;
  };
  u16*   wqkv  = (u16*)alloc((size_t)3 * D * D * 2);
  u16*   wout  = (u16*)alloc((size_t)D * D * 2);
  u16*   wfc1  = (u16*)alloc((size_t)DFF * D * 2);
  u16*   wfc2  = (u16*)alloc((size_t)D * DFF * 2);
  u16*   hbf   = (u16*)alloc((size_t)T * D * 2);
  u16*   qkvbf = (u16*)alloc((size_t)T * 3 * D * 2);
  u16*   qbf   = (u16*)alloc((size_t)T * D * 2);
  u16*   kbf   = (u16*)alloc((size_t)T * D * 2);
  u16*   vtbf  = (u16*)alloc((size_t)T * D * 2);
  u16*   obf   = (u16*)alloc((size_t)T * D * 2);
  float* x1    = (float*)alloc((size_t)T * D * 4);
  u16*   ffbf  = (u16*)alloc((size_t)T * DFF * 2);

  // weights -> bf16
  cvt_f32_bf16<<<4096, 256, 0, stream>>>(qkv_w, wqkv, (size_t)3 * D * D);
  cvt_f32_bf16<<<4096, 256, 0, stream>>>(out_w, wout, (size_t)D * D);
  cvt_f32_bf16<<<4096, 256, 0, stream>>>(fc1_w, wfc1, (size_t)DFF * D);
  cvt_f32_bf16<<<4096, 256, 0, stream>>>(fc2_w, wfc2, (size_t)D * DFF);

  // attention branch
  rmsnorm_to_bf16<<<T, 256, 0, stream>>>(x, ln1_w, hbf, D);
  gemm_bf16<0><<<dim3(3 * D / 128, T / 128), 256, 0, stream>>>(
      hbf, wqkv, qkv_b, nullptr, qkvbf, T, 3 * D, D);
  rope_split<<<(T * D) / 256, 256, 0, stream>>>(qkvbf, qbf, kbf, vtbf);
  attn_flash<<<(B * 16 * (S / 16)) / 8, 256, 0, stream>>>(qbf, kbf, vtbf, obf);
  gemm_bf16<2><<<dim3(D / 128, T / 128), 256, 0, stream>>>(
      obf, wout, out_b, x, x1, T, D, D);

  // FFN branch
  rmsnorm_to_bf16<<<T, 256, 0, stream>>>(x1, ln2_w, hbf, D);
  gemm_bf16<1><<<dim3(DFF / 128, T / 128), 256, 0, stream>>>(
      hbf, wfc1, fc1_b, nullptr, ffbf, T, DFF, D);
  gemm_bf16<2><<<dim3(D / 128, T / 128), 256, 0, stream>>>(
      ffbf, wfc2, fc2_b, x1, (float*)d_out, T, D, DFF);
}
